// TransformerBlock_24567212933407
// MI455X (gfx1250) — compile-verified, matrix operand order checked by reading, not run on hardware
//
#include <hip/hip_runtime.h>
#include <stdint.h>

// ---------------------------------------------------------------------------
// Problem constants (from reference): B=4, C=192, H=W=256, BS=16
// windows: 16x16 per image -> 1024 windows, N=256 tokens/window
// ---------------------------------------------------------------------------
#define C_DIM   192
#define N_TOK   256
#define HW      65536          // H*W
#define NWIN    1024           // B * 16 * 16
#define O_DIM   576            // 3*C

typedef __attribute__((ext_vector_type(16))) __bf16 v16bf;
typedef __attribute__((ext_vector_type(8)))  float  v8f;
typedef __attribute__((ext_vector_type(4)))  uint32_t u32x4;
typedef __attribute__((ext_vector_type(8)))  uint32_t u32x8;

union Frag {
    v16bf    v;
    uint16_t u[16];
    uint4    q[2];
};

union Pack8 {
    uint16_t u[8];
    uint4    q;
};

// Native f32 -> bf16 (RNE fptrunc; backend selects hardware cvt on gfx1250).
__device__ __forceinline__ uint16_t f2bf(float x) {
    union { __bf16 h; uint16_t u; } c;
    c.h = (__bf16)x;
    return c.u;
}

__device__ __forceinline__ v8f wmma_bf16(v16bf a, v16bf b, v8f c) {
    // v_wmma_f32_16x16x32_bf16: D = A(16x32) x B(32x16) + C(16x16 f32)
    return __builtin_amdgcn_wmma_f32_16x16x32_bf16(
        /*neg_a=*/false, a, /*neg_b=*/false, b,
        /*c_mod=*/(short)0, c, /*reuse_a=*/false, /*reuse_b=*/false);
}

// Async copy 16B global -> LDS (tracked on ASYNCcnt). Each lane supplies its
// own LDS byte offset (VDST) and global address (VADDR). [CDNA5]
__device__ __forceinline__ void async_g2l_b128(uint32_t lds_byte_off,
                                               const void* gsrc) {
    asm volatile("global_load_async_to_lds_b128 %0, %1, off"
                 :: "v"(lds_byte_off), "v"(gsrc)
                 : "memory");
}

__device__ __forceinline__ void wait_asynccnt0() {
    asm volatile("s_wait_asynccnt 0x0" ::: "memory");
}

// Tensor Data Mover: DMA a linear block of `n_qwords` 8-byte elements from
// global memory to LDS with one instruction (tracked on TENSORcnt). D# built
// per CDNA5 ISA: group0 {count=1, lds_addr, global_addr,
// type=2}; group1 {data_size=8B, tensor_dim0=tile_dim0=n_qwords, dim1=1}.
// Issue from ONE wave only; wait with s_wait_tensorcnt 0.
__device__ __forceinline__ void tdm_load_linear(uint32_t lds_byte_off,
                                                const void* gsrc,
                                                uint32_t n_qwords) {
    uint64_t ga = (uint64_t)(uintptr_t)gsrc;
    u32x4 g0;
    g0[0] = 1u;                                            // count=1, user mode
    g0[1] = lds_byte_off;                                  // lds_addr
    g0[2] = (uint32_t)ga;                                  // global_addr[31:0]
    g0[3] = (uint32_t)((ga >> 32) & 0x01FFFFFFu) | (2u << 30); // addr[56:32]|type=2
    u32x8 g1;
    g1[0] = (3u << 16);                                    // data_size=3 (8B)
    g1[1] = (n_qwords & 0xFFFFu) << 16;                    // tensor_dim0[15:0]
    g1[2] = ((n_qwords >> 16) & 0xFFFFu) | (1u << 16);     // td0[31:16], tensor_dim1=1
    g1[3] = (n_qwords & 0xFFFFu) << 16;                    // tile_dim0
    g1[4] = 0u;                                            // tile_dim1/2 unused
    g1[5] = n_qwords;                                      // tensor_dim0_stride lo
    g1[6] = 0u;
    g1[7] = 0u;
    asm volatile("tensor_load_to_lds %0, %1"
                 :: "s"(g0), "s"(g1)
                 : "memory");
}

// Load a 16x32 bf16 fragment (A-operand pattern, also valid for B when the
// memory layout is [col][row]) from row-major memory with leading dim `ld`.
// ISA layout (16-bit A 16x32): lane = (g = lane/16, m = lane%16) holds row m,
// K = g*8 + {0..7} in VGPRs 0-3 and K = g*8 + 16 + {0..7} in VGPRs 4-7.
__device__ __forceinline__ v16bf load_frag(const uint16_t* __restrict__ base,
                                           int ld, int kb) {
    int lane = threadIdx.x & 31;
    int g = lane >> 4, m = lane & 15;
    const uint16_t* p = base + (size_t)m * ld + kb + g * 8;
    Frag f;
    f.q[0] = *(const uint4*)(p);
    f.q[1] = *(const uint4*)(p + 16);
    return f.v;
}

// Same pattern but from LDS.
__device__ __forceinline__ v16bf load_frag_lds(const uint16_t* base,
                                               int ld, int kb) {
    int lane = threadIdx.x & 31;
    int g = lane >> 4, m = lane & 15;
    const uint16_t* p = base + m * ld + kb + g * 8;
    Frag f;
    f.q[0] = *(const uint4*)(p);
    f.q[1] = *(const uint4*)(p + 16);
    return f.v;
}

__device__ __forceinline__ v8f vzero8() {
    v8f z = {0.f, 0.f, 0.f, 0.f, 0.f, 0.f, 0.f, 0.f};
    return z;
}

// ---------------------------------------------------------------------------
// Kernel 0: convert fp32 weights to bf16 once.
// ---------------------------------------------------------------------------
__global__ __launch_bounds__(256) void cvt_weights_kernel(
    const float* __restrict__ qkv_w, const float* __restrict__ proj_w,
    uint16_t* __restrict__ qkv_w_bf, uint16_t* __restrict__ proj_w_bf)
{
    int i = blockIdx.x * 256 + threadIdx.x;
    if (i < O_DIM * C_DIM) qkv_w_bf[i] = f2bf(qkv_w[i]);
    if (i < C_DIM * C_DIM) proj_w_bf[i] = f2bf(proj_w[i]);
}

// ---------------------------------------------------------------------------
// Kernel 1: QKV projection, one window per workgroup (256 thr = 8 waves).
//   LDS: xT [N][C] bf16 (96KB, staged+transposed by VALU) followed by the
//   full qkv weight matrix [576][192] bf16 (216KB) DMA'd in by the Tensor
//   Data Mover concurrently with the x staging. Total 319488 B <= 320KB.
//   Q, K stored token-major [win][N][C] bf16; V stored [win][C][N] bf16.
//   Inner GEMM loop software-pipelines B-fragment ds loads one WMMA ahead.
// ---------------------------------------------------------------------------
__global__ __launch_bounds__(256) void qkv_kernel(
    const float* __restrict__ x, const uint16_t* __restrict__ w_bf,
    const float* __restrict__ qkv_b,
    uint16_t* __restrict__ qT, uint16_t* __restrict__ kT,
    uint16_t* __restrict__ vM)
{
    extern __shared__ unsigned char smem[];
    uint16_t* xT   = (uint16_t*)smem;                      // [N_TOK][C_DIM]
    uint16_t* wlds = (uint16_t*)(smem + N_TOK * C_DIM * 2);// [O_DIM][C_DIM]

    const int bwin = blockIdx.x;
    const int b  = bwin >> 8;
    const int wn = bwin & 255;
    const int hi = wn >> 4, wi = wn & 15;
    const int tid = threadIdx.x;
    const int wv  = tid >> 5;

    // ---- kick off TDM DMA of the whole qkv weight matrix into LDS ----
    if (wv == 0) {
        tdm_load_linear((uint32_t)(uintptr_t)wlds, w_bf,
                        (O_DIM * C_DIM * 2) / 8);          // 27648 qwords
    }

    // ---- concurrently stage x window into LDS, transposed, bf16 ----
    const float* xw = x + (size_t)b * C_DIM * HW + hi * 16 * 256 + wi * 16;
    for (int f4 = tid; f4 < C_DIM * N_TOK / 4; f4 += 256) {
        int f = f4 * 4;
        int c = f >> 8;          // / 256
        int n = f & 255;
        int r = n >> 4, cc = n & 15;
        float4 v = *(const float4*)(xw + (size_t)c * HW + r * 256 + cc);
        xT[(n + 0) * C_DIM + c] = f2bf(v.x);
        xT[(n + 1) * C_DIM + c] = f2bf(v.y);
        xT[(n + 2) * C_DIM + c] = f2bf(v.z);
        xT[(n + 3) * C_DIM + c] = f2bf(v.w);
    }
    if (wv == 0) __builtin_amdgcn_s_wait_tensorcnt(0);
    __syncthreads();

    const int lane = tid & 31;
    const int g    = lane >> 4;
    const int m15  = lane & 15;

    // 36 output tiles of 16 rows (576 = 3*192), 8 waves round-robin.
    for (int ot = wv; ot < 36; ot += 8) {
        const int o0 = ot * 16;

        v8f acc[16];
        for (int i = 0; i < 16; ++i) acc[i] = vzero8();

        for (int kb = 0; kb < C_DIM; kb += 32) {
            v16bf afr  = load_frag_lds(wlds + o0 * C_DIM, C_DIM, kb);
            v16bf bcur = load_frag_lds(xT, C_DIM, kb);
            #pragma unroll
            for (int nt = 0; nt < 16; ++nt) {
                v16bf bnext = (nt < 15)
                    ? load_frag_lds(xT + (nt + 1) * 16 * C_DIM, C_DIM, kb)
                    : bcur;
                acc[nt] = wmma_bf16(afr, bcur, acc[nt]);
                bcur = bnext;
            }
        }

        // bias for this tile's 8 output rows (o = o0 + 8*g + r)
        float bias[8];
        for (int r = 0; r < 8; ++r) bias[r] = qkv_b[o0 + 8 * g + r];

        if (ot < 24) {
            // Q or K: store transposed (token-major), packed b128 per lane.
            uint16_t* dst = (ot < 12) ? qT : kT;
            const int oo0 = (ot % 12) * 16;
            const size_t base = (size_t)bwin * N_TOK * C_DIM;
            for (int nt = 0; nt < 16; ++nt) {
                int n = nt * 16 + m15;
                Pack8 pk;
                for (int r = 0; r < 8; ++r)
                    pk.u[r] = f2bf(acc[nt][r] + bias[r]);
                *(uint4*)(dst + base + (size_t)n * C_DIM + oo0 + 8 * g) = pk.q;
            }
        } else {
            // V: store [C][N]
            const int oo0 = (ot - 24) * 16;
            const size_t base = (size_t)bwin * C_DIM * N_TOK;
            for (int nt = 0; nt < 16; ++nt) {
                int n = nt * 16 + m15;
                for (int r = 0; r < 8; ++r) {
                    int c = oo0 + 8 * g + r;
                    vM[base + (size_t)c * N_TOK + n] = f2bf(acc[nt][r] + bias[r]);
                }
            }
        }
    }
}

// ---------------------------------------------------------------------------
// Kernel 2: attention, one window per workgroup (256 thr = 8 waves).
//   Phase 1: S = Q^T K  -> LDS f32 [256][260] (pad col 256 holds 1/rowsum)
//   Phase 2: row softmax, one row per thread, float4-vectorized
//   Phase 3: out^T = att x V^T, written channels-last y[b][p][C] bf16
//   Global B-fragment streams are software-pipelined two WMMAs deep.
// Dynamic LDS: 256*260*4 = 266240 bytes (fits 320KB WGP LDS).
// ---------------------------------------------------------------------------
__global__ __launch_bounds__(256) void attn_kernel(
    const uint16_t* __restrict__ qT, const uint16_t* __restrict__ kT,
    const uint16_t* __restrict__ vM, uint16_t* __restrict__ y)
{
    extern __shared__ unsigned char smem[];
    float* S = (float*)smem;                 // [256][260]
    const int LDW = 260;

    const int bwin = blockIdx.x;
    const int b  = bwin >> 8;
    const int wn = bwin & 255;
    const int hi = wn >> 4, wi = wn & 15;

    const int tid  = threadIdx.x;
    const int wv   = tid >> 5;
    const int lane = tid & 31;
    const int g    = lane >> 4;
    const int m15  = lane & 15;

    const uint16_t* q = qT + (size_t)bwin * N_TOK * C_DIM;
    const uint16_t* k = kT + (size_t)bwin * N_TOK * C_DIM;
    const uint16_t* v = vM + (size_t)bwin * C_DIM * N_TOK;

    // ---- phase 1: scores ----
    for (int nh = 0; nh < 2; ++nh) {
        const int nt = wv + nh * 8;
        v16bf afr[6];
        for (int kc = 0; kc < 6; ++kc)
            afr[kc] = load_frag(q + (size_t)nt * 16 * C_DIM, C_DIM, kc * 32);
        for (int mt = 0; mt < 16; ++mt) {
            const uint16_t* kb = k + (size_t)mt * 16 * C_DIM;
            v8f acc = vzero8();
            v16bf bcur  = load_frag(kb, C_DIM, 0);
            v16bf bnext = load_frag(kb, C_DIM, 32);
            #pragma unroll
            for (int kc = 0; kc < 6; ++kc) {
                v16bf bfar = (kc < 4) ? load_frag(kb, C_DIM, (kc + 2) * 32)
                                      : bcur;
                acc = wmma_bf16(afr[kc], bcur, acc);
                bcur = bnext;
                bnext = bfar;
            }
            const int mcol = mt * 16 + m15;
            for (int r = 0; r < 8; ++r) {
                int n = nt * 16 + r + 8 * g;
                S[n * LDW + mcol] = acc[r];
            }
        }
    }
    __syncthreads();

    // ---- phase 2: softmax (row `tid`), float4 vectorized ----
    {
        float* row = S + tid * LDW;
        float4* r4 = (float4*)row;           // row base is 16B aligned (1040B stride)
        float mx = -3.4e38f;
        for (int i = 0; i < 64; ++i) {
            float4 t = r4[i];
            mx = fmaxf(mx, fmaxf(fmaxf(t.x, t.y), fmaxf(t.z, t.w)));
        }
        float sum = 0.f;
        for (int i = 0; i < 64; ++i) {
            float4 t = r4[i];
            t.x = __expf(t.x - mx); t.y = __expf(t.y - mx);
            t.z = __expf(t.z - mx); t.w = __expf(t.w - mx);
            r4[i] = t;
            sum += (t.x + t.y) + (t.z + t.w);
        }
        row[256] = 1.0f / sum;   // fold normalization into phase 3
    }
    __syncthreads();

    // ---- phase 3: out^T = att x V^T ----
    for (int nh = 0; nh < 2; ++nh) {
        const int nt = wv + nh * 8;
        // att A-fragments for full K=256 (8 chunks of 32), normalized on the fly
        v16bf att[8];
        {
            const int n = nt * 16 + m15;       // this lane's A row
            const float* row = S + n * LDW;
            const float rs = row[256];
            const float4* r4 = (const float4*)row;
            for (int kc = 0; kc < 8; ++kc) {
                Frag f;
                const int q0 = (kc * 32 + g * 8) >> 2;   // float4 index
                float4 a0 = r4[q0 + 0], a1 = r4[q0 + 1]; // m = mb .. mb+7
                float4 b0 = r4[q0 + 4], b1 = r4[q0 + 5]; // m = mb+16 .. mb+23
                f.u[0]  = f2bf(a0.x * rs); f.u[1]  = f2bf(a0.y * rs);
                f.u[2]  = f2bf(a0.z * rs); f.u[3]  = f2bf(a0.w * rs);
                f.u[4]  = f2bf(a1.x * rs); f.u[5]  = f2bf(a1.y * rs);
                f.u[6]  = f2bf(a1.z * rs); f.u[7]  = f2bf(a1.w * rs);
                f.u[8]  = f2bf(b0.x * rs); f.u[9]  = f2bf(b0.y * rs);
                f.u[10] = f2bf(b0.z * rs); f.u[11] = f2bf(b0.w * rs);
                f.u[12] = f2bf(b1.x * rs); f.u[13] = f2bf(b1.y * rs);
                f.u[14] = f2bf(b1.z * rs); f.u[15] = f2bf(b1.w * rs);
                att[kc] = f.v;
            }
        }
        for (int ct = 0; ct < 12; ++ct) {
            const uint16_t* vb = v + (size_t)ct * 16 * N_TOK;
            v8f acc = vzero8();
            v16bf bcur  = load_frag(vb, N_TOK, 0);
            v16bf bnext = load_frag(vb, N_TOK, 32);
            #pragma unroll
            for (int kc = 0; kc < 8; ++kc) {
                v16bf bfar = (kc < 6) ? load_frag(vb, N_TOK, (kc + 2) * 32)
                                      : bcur;
                acc = wmma_bf16(att[kc], bcur, acc);
                bcur = bnext;
                bnext = bfar;
            }
            // store y channels-last: y[b][p][c]
            const int cc = ct * 16 + m15;
            for (int r = 0; r < 8; ++r) {
                int n = nt * 16 + r + 8 * g;
                int p = (hi * 16 + (n >> 4)) * 256 + wi * 16 + (n & 15);
                y[((size_t)b * HW + p) * C_DIM + cc] = f2bf(acc[r]);
            }
        }
    }
}

// ---------------------------------------------------------------------------
// Kernel 3: output projection + bias + residual.
//   out[b][o][p] = sum_c W[o][c] * y[b][p][c] + proj_b[o] + x[b][o][p]
//   proj_w staged in LDS via async global->LDS b128 copies (ASYNCcnt path).
//   Each wave owns one p-tile, loads its 6 B-fragments once and reuses them
//   across all 12 o-tiles; A-fragment ds loads pipelined one WMMA ahead.
//   Streaming loads/stores marked non-temporal.
// Grid: 4 batches * 512 blocks (8 p-tiles each) = 2048 blocks.
// ---------------------------------------------------------------------------
__global__ __launch_bounds__(256) void proj_kernel(
    const uint16_t* __restrict__ y, const uint16_t* __restrict__ w_bf,
    const float* __restrict__ proj_b, const float* __restrict__ x,
    float* __restrict__ out)
{
    extern __shared__ unsigned char smem[];
    uint16_t* wl = (uint16_t*)smem;          // [192][192] bf16

    const int tid = threadIdx.x;

    // ---- async-stage proj weights into LDS (73728 B, 16B per op) ----
    {
        const uint32_t lds_base = (uint32_t)(uintptr_t)wl;  // addr[31:0] = LDS offset
        for (int i = tid; i < C_DIM * C_DIM / 8; i += 256)
            async_g2l_b128(lds_base + (uint32_t)i * 16, w_bf + (size_t)i * 8);
        wait_asynccnt0();
    }
    __syncthreads();

    const int wv   = tid >> 5;
    const int lane = tid & 31;
    const int g    = lane >> 4;
    const int m15  = lane & 15;

    const int b  = blockIdx.x >> 9;          // / 512
    const int pb = blockIdx.x & 511;
    const int p0 = (pb * 8 + wv) * 16;       // this wave's p-tile base

    const uint16_t* yb = y + (size_t)b * HW * C_DIM;

    // B-fragments: B[k=c][col=p] from y[p][c] -> contiguous in c
    v16bf bfr[6];
    for (int kc = 0; kc < 6; ++kc)
        bfr[kc] = load_frag(yb + (size_t)p0 * C_DIM, C_DIM, kc * 32);

    const int p = p0 + m15;
    for (int ot = 0; ot < 12; ++ot) {
        const int o0 = ot * 16;
        v8f acc = vzero8();
        v16bf acur = load_frag_lds(wl + o0 * C_DIM, C_DIM, 0);
        #pragma unroll
        for (int kc = 0; kc < 6; ++kc) {
            v16bf anext = (kc < 5)
                ? load_frag_lds(wl + o0 * C_DIM, C_DIM, (kc + 1) * 32)
                : acur;
            acc = wmma_bf16(acur, bfr[kc], acc);
            acur = anext;
        }
        for (int r = 0; r < 8; ++r) {
            int o = o0 + 8 * g + r;
            size_t idx = ((size_t)(b * C_DIM + o)) * HW + p;
            float xr  = __builtin_nontemporal_load(x + idx);
            float val = acc[r] + proj_b[o] + xr;
            __builtin_nontemporal_store(val, out + idx);
        }
    }
}

// ---------------------------------------------------------------------------
// Host launcher
// ---------------------------------------------------------------------------
extern "C" void kernel_launch(void* const* d_in, const int* in_sizes, int n_in,
                              void* d_out, int out_size, void* d_ws, size_t ws_size,
                              hipStream_t stream) {
    const float* x      = (const float*)d_in[0];
    const float* qkv_w  = (const float*)d_in[1];
    const float* qkv_b  = (const float*)d_in[2];
    const float* proj_w = (const float*)d_in[3];
    const float* proj_b = (const float*)d_in[4];
    float* out = (float*)d_out;

    uint8_t* ws = (uint8_t*)d_ws;
    const size_t SZ_QKVW = (size_t)O_DIM * C_DIM * 2;          // 221184
    const size_t SZ_PRJW = (size_t)C_DIM * C_DIM * 2;          //  73728
    const size_t SZ_MAT  = (size_t)NWIN * N_TOK * C_DIM * 2;   // ~96 MB each

    size_t off = 0;
    uint16_t* qkvw_bf = (uint16_t*)(ws + off); off += SZ_QKVW;
    uint16_t* projw_bf = (uint16_t*)(ws + off); off += SZ_PRJW;
    off = (off + 255) & ~(size_t)255;
    uint16_t* qT = (uint16_t*)(ws + off); off += SZ_MAT;
    uint16_t* kT = (uint16_t*)(ws + off); off += SZ_MAT;
    uint16_t* vM = (uint16_t*)(ws + off); off += SZ_MAT;
    uint16_t* y  = (uint16_t*)(ws + off); off += SZ_MAT;

    cvt_weights_kernel<<<(O_DIM * C_DIM + 255) / 256, 256, 0, stream>>>(
        qkv_w, proj_w, qkvw_bf, projw_bf);

    // LDS: xT (98304 B) + full qkv weights (221184 B) = 319488 B <= 320KB
    qkv_kernel<<<NWIN, 256, N_TOK * C_DIM * 2 + O_DIM * C_DIM * 2, stream>>>(
        x, qkvw_bf, qkv_b, qT, kT, vM);

    attn_kernel<<<NWIN, 256, 256 * 260 * 4, stream>>>(qT, kT, vM, y);

    proj_kernel<<<2048, 256, C_DIM * C_DIM * 2, stream>>>(
        y, projw_bf, proj_b, x, out);
}
